// MambaLayer_52080773431740
// MI455X (gfx1250) — matrix operand order repaired and checked by reading, not compiled, not run on hardware
//
#include <hip/hip_runtime.h>
#include <hip/hip_bf16.h>

// ---------------- problem constants (from reference) ----------------
#define DIMC     256
#define D_STATE  16
#define D_CONV   4
#define D_INNER  512
#define HALFC    256
#define DT_RANK  16
#define BATCH    2
#define NTOK     2304     // 48*48
#define SCAN_T   32       // timesteps staged per async LDS chunk

typedef __attribute__((ext_vector_type(16))) __bf16 v16bf;
typedef __attribute__((ext_vector_type(8)))  float  v8f;

__device__ __forceinline__ float siluf(float x) {
    return x / (1.0f + __expf(-x));
}

__device__ __forceinline__ float block_reduce_sum(float v, float* sbuf) {
    int tid = threadIdx.x;
    sbuf[tid] = v;
    __syncthreads();
    for (int s = blockDim.x >> 1; s > 0; s >>= 1) {
        if (tid < s) sbuf[tid] += sbuf[tid + s];
        __syncthreads();
    }
    float r = sbuf[0];
    __syncthreads();
    return r;
}

// ---------------- WMMA GEMM: C[M,N] = A[M,K] * B, optional residual ----------
// B_IS_NK == true : B stored [N,K] row-major (token-major activations)
// B_IS_NK == false: B stored [K,N] row-major (channel-major (d,l) tensors)
// MT: number of stacked 16-row M tiles per wave (shares one B fragment).
// A is unbatched weights; B/C/Resid batched via strides, blockIdx.z = batch.
// fp32 -> bf16 conversion uses native v_cvt (RNE) via plain __bf16 casts.
template<bool B_IS_NK, bool HAS_RESID, int MT>
__global__ void wmma_gemm(const float* __restrict__ A,
                          const float* __restrict__ Bmat,
                          const float* __restrict__ Resid,
                          float* __restrict__ C,
                          int M, int Ncols, int K,
                          long bStrideB, long bStrideC) {
    const int lane = threadIdx.x;          // 0..31, wave32
    const int half = lane >> 4;            // 0 | 1
    const int sub  = lane & 15;
    const int tileM = blockIdx.y * (16 * MT);
    const int tileN = (blockIdx.x * blockDim.y + threadIdx.y) * 16;
    if (tileN >= Ncols) return;            // wave-uniform, EXEC stays full

    const float* Bp = Bmat + (long)blockIdx.z * bStrideB;
    float*       Cp = C    + (long)blockIdx.z * bStrideC;
    const float* Rp = HAS_RESID ? (Resid + (long)blockIdx.z * bStrideC) : nullptr;

    const v8f vzero = {};
    v8f acc[MT];
    #pragma unroll
    for (int m = 0; m < MT; ++m) acc[m] = vzero;

    const int bn = tileN + sub;            // B col / C col for this lane

    for (int k0 = 0; k0 < K; k0 += 32) {
        if (B_IS_NK) __builtin_prefetch(&Bp[(long)bn * K + k0 + 64], 0, 0);

        // B fragment: contiguous k order, same as A. elem e -> K = e + 16*half
        v16bf ub;
        #pragma unroll
        for (int e = 0; e < 16; ++e) {
            int kk = k0 + e + 16 * half;
            float f = B_IS_NK ? Bp[(long)bn * K + kk]
                              : Bp[(long)kk * Ncols + bn];
            ub[e] = (__bf16)f;
        }
        // A fragments (one per stacked M tile) reuse the B fragment
        #pragma unroll
        for (int m = 0; m < MT; ++m) {
            const int am = tileM + m * 16 + sub;
            v16bf ua;
            #pragma unroll
            for (int e = 0; e < 16; ++e) {
                int kk = k0 + e + 16 * half;
                ua[e] = (__bf16)A[(long)am * K + kk];
            }
            acc[m] = __builtin_amdgcn_wmma_f32_16x16x32_bf16(
                         false, ua, false, ub, (short)0, acc[m], false, false);
        }
    }
    // C/D f32 16x16 layout: VGPR r holds row M = r + 8*half, col = sub
    #pragma unroll
    for (int m = 0; m < MT; ++m) {
        #pragma unroll
        for (int r = 0; r < 8; ++r) {
            int mm = tileM + m * 16 + 8 * half + r;
            long off = (long)mm * Ncols + bn;
            float v = acc[m][r];
            if (HAS_RESID) v += Rp[off];
            Cp[off] = v;
        }
    }
}

// ---------------- LayerNorm over channel dim, tokenizing (B,C,N)->(B,N,C) ----
__global__ void layernorm_tok(const float* __restrict__ x,
                              const float* __restrict__ g,
                              const float* __restrict__ bta,
                              float* __restrict__ xn) {
    __shared__ float sbuf[DIMC];
    int token = blockIdx.x;                 // b*NTOK + n
    int b = token / NTOK, n = token % NTOK;
    int d = threadIdx.x;                    // 256 threads
    float v = x[((long)b * DIMC + d) * NTOK + n];
    float mu = block_reduce_sum(v, sbuf) * (1.0f / DIMC);
    float dv = v - mu;
    float var = block_reduce_sum(dv * dv, sbuf) * (1.0f / DIMC);
    float r = rsqrtf(var + 1e-5f);
    xn[(long)token * DIMC + d] = dv * r * g[d] + bta[d];
}

// ---------------- depthwise conv1d (4-tap) + SiLU, optional sequence flip ----
__global__ void dwconv_silu(const float* __restrict__ in,
                            float* __restrict__ out,
                            const float* __restrict__ w,
                            const float* __restrict__ bias,
                            int C, long inBatchStride, long outBatchStride,
                            int padLeft, int rev, int total) {
    for (int idx = blockIdx.x * blockDim.x + threadIdx.x; idx < total;
         idx += gridDim.x * blockDim.x) {
        int t = idx % NTOK;
        int c = (idx / NTOK) % C;
        int b = idx / (NTOK * C);
        const float* xin = in + (long)b * inBatchStride + (long)c * NTOK;
        float acc = bias[c];
        #pragma unroll
        for (int j = 0; j < D_CONV; ++j) {
            int tt = t + j - padLeft;
            float v = 0.0f;
            if (tt >= 0 && tt < NTOK) v = rev ? xin[NTOK - 1 - tt] : xin[tt];
            acc += w[c * D_CONV + j] * v;
        }
        out[(long)b * outBatchStride + (long)c * NTOK + t] = siluf(acc);
    }
}

// ---------------- token importance score (dt-rank path) ---------------------
__global__ void score_kernel(const float* __restrict__ xc,     // (B,256,N)
                             const float* __restrict__ xproj_w,// (48,256)
                             const float* __restrict__ dtw,    // (256,16)
                             const float* __restrict__ dtb,    // (256)
                             const float* __restrict__ imp,    // (256)
                             float* __restrict__ scores) {
    __shared__ float sbuf[HALFC];
    int token = blockIdx.x;
    int b = token / NTOK, l = token % NTOK;
    int d = threadIdx.x;
    float xv = xc[((long)b * HALFC + d) * NTOK + l];
    float dtl = dtb[d];
    for (int r = 0; r < DT_RANK; ++r) {
        float dtr = block_reduce_sum(xv * xproj_w[r * HALFC + d], sbuf);
        dtl += dtr * dtw[d * DT_RANK + r];
    }
    float sc = block_reduce_sum(dtl * imp[d], sbuf);
    if (d == 0) scores[token] = sc;
}

// ---------------- soft rank: sr_i = 0.5 + sum_j sigmoid(s_i - s_j) ----------
__global__ void softrank_kernel(const float* __restrict__ scores,
                                float* __restrict__ sr) {
    __shared__ float sbuf[256];
    int token = blockIdx.x;
    int b = token / NTOK, i = token % NTOK;
    float si = scores[b * NTOK + i];
    float sum = 0.0f;
    for (int j = threadIdx.x; j < NTOK; j += 256) {
        float dj = si - scores[b * NTOK + j];
        sum += 1.0f / (1.0f + __expf(-dj));
    }
    float tot = block_reduce_sum(sum, sbuf);
    if (threadIdx.x == 0) sr[token] = 0.5f + tot;
}

__global__ void zero_kernel(float* __restrict__ p, int total) {
    for (int i = blockIdx.x * blockDim.x + threadIdx.x; i < total;
         i += gridDim.x * blockDim.x) p[i] = 0.0f;
}

// ---------------- fractional scatter reorder --------------------------------
__global__ void scatter_kernel(const float* __restrict__ xn,
                               const float* __restrict__ sr,
                               float* __restrict__ xs) {
    int token = blockIdx.x;
    int b = token / NTOK, i = token % NTOK;
    float s = sr[token];
    int rf = (int)floorf(s);
    rf = min(max(rf, 0), NTOK - 1);
    float frac = s - (float)rf;
    int i1 = min(rf + 1, NTOK - 1);
    float m = (rf + 1 < NTOK) ? 1.0f : 0.0f;
    int d = threadIdx.x;
    float v = xn[((long)b * NTOK + i) * DIMC + d];
    atomicAdd(&xs[((long)b * NTOK + rf) * DIMC + d], v * (1.0f - frac));
    atomicAdd(&xs[((long)b * NTOK + i1) * DIMC + d], v * frac * m);
}

// ---------------- delta = softplus(dt @ dtproj_w^T + b) ---------------------
__global__ void delta_kernel(const float* __restrict__ xd,   // (B,48,N)
                             const float* __restrict__ dtw,  // (512,16)
                             const float* __restrict__ dtb,  // (512)
                             float* __restrict__ delta, int total) {
    for (int idx = blockIdx.x * blockDim.x + threadIdx.x; idx < total;
         idx += gridDim.x * blockDim.x) {
        int t = idx % NTOK;
        int d = (idx / NTOK) % D_INNER;
        int b = idx / (NTOK * D_INNER);
        float acc = dtb[d];
        #pragma unroll
        for (int r = 0; r < DT_RANK; ++r)
            acc += xd[((long)b * 48 + r) * NTOK + t] * dtw[d * DT_RANK + r];
        float sp = (acc > 20.0f) ? acc : log1pf(__expf(acc));
        delta[idx] = sp;
    }
}

// ---------------- selective scan (sequential in t, parallel over b,d,n) -----
// block = 256 threads = 16 d-channels x 16 states. n-group reduce via shfl_xor.
// Streams (delta, u, B, C) are staged in 32-step chunks into LDS with CDNA5
// async-to-LDS loads, double-buffered: chunk c+1 is in flight (ASYNCcnt=2 per
// wave) while chunk c is consumed from LDS.
__global__ void scan_kernel(const float* __restrict__ delta,  // (B,512,N)
                            const float* __restrict__ u,      // (B,512,N) conv'd x
                            const float* __restrict__ xd,     // (B,48,N) Bm=16..31 Cm=32..47
                            const float* __restrict__ xz2,    // (B,1024,N) for z gate
                            const float* __restrict__ A_log,  // (512,16)
                            const float* __restrict__ Dp,     // (512)
                            float* __restrict__ y2,           // (B,512,N)
                            int rev) {
    extern __shared__ float smem[];        // 2 bufs * 4 streams * 16 rows * 32 = 4096 floats
    const int b    = blockIdx.z;
    const int tid  = threadIdx.x;
    const int n    = tid & 15;
    const int dloc = tid >> 4;
    const int d    = blockIdx.y * 16 + dloc;

    const float Aval = -__expf(A_log[d * D_STATE + n]);
    const float Dd   = Dp[d];

    const float* z_row = xz2 + ((long)b * 2 * D_INNER + D_INNER + d) * NTOK;
    float*       y_row = y2  + ((long)b * D_INNER + d) * NTOK;

    // stage one 4-stream x 16-row x 32-step chunk: 512 b128 segments / block,
    // 2 per thread.  seg -> (stream a, row, 4-float segment s)
    auto issue_chunk = [&](int t0, int buf) {
        #pragma unroll
        for (int i = 0; i < 2; ++i) {
            int seg = tid + 256 * i;
            int a   = seg >> 7;            // 0..3
            int row = (seg >> 3) & 15;
            int s   = seg & 7;
            const float* g;
            if (a == 0)      g = delta + ((long)b * D_INNER + blockIdx.y * 16 + row) * NTOK;
            else if (a == 1) g = u     + ((long)b * D_INNER + blockIdx.y * 16 + row) * NTOK;
            else if (a == 2) g = xd + ((long)b * 48 + DT_RANK + row) * NTOK;
            else             g = xd + ((long)b * 48 + DT_RANK + D_STATE + row) * NTOK;
            g += t0 + 4 * s;
            unsigned lds = (unsigned)((buf * 2048 + a * 512 + row * 32 + s * 4) * 4);
            asm volatile("global_load_async_to_lds_b128 %0, %1, off"
                         :: "v"(lds), "v"(g) : "memory");
        }
    };

    float h = 0.0f;
    int buf = 0;
    issue_chunk(0, 0);
    for (int t0 = 0; t0 < NTOK; t0 += SCAN_T) {
        if (t0 + SCAN_T < NTOK) {
            issue_chunk(t0 + SCAN_T, buf ^ 1);
            asm volatile("s_wait_asynccnt 2" ::: "memory");  // current chunk landed
        } else {
            asm volatile("s_wait_asynccnt 0" ::: "memory");
        }
        __syncthreads();
        const float* sd = smem + buf * 2048 +    0 + dloc * 32;
        const float* su = smem + buf * 2048 +  512 + dloc * 32;
        const float* sB = smem + buf * 2048 + 1024 + n * 32;
        const float* sC = smem + buf * 2048 + 1536 + n * 32;
        #pragma unroll 4
        for (int tt = 0; tt < SCAN_T; ++tt) {
            float dl = sd[tt];
            float uv = su[tt];
            float da = __expf(dl * Aval);
            h = da * h + dl * sB[tt] * uv;
            float part = h * sC[tt];
            part += __shfl_xor(part, 1, 16);
            part += __shfl_xor(part, 2, 16);
            part += __shfl_xor(part, 4, 16);
            part += __shfl_xor(part, 8, 16);
            if (n == 0) {
                int t = t0 + tt;
                int l = rev ? (NTOK - 1 - t) : t;
                float val = (part + Dd * uv) * siluf(z_row[l]);
                if (rev) y_row[l] += val; else y_row[l] = val;
            }
        }
        __syncthreads();
        buf ^= 1;
    }
}

// ============================== launcher =====================================
extern "C" void kernel_launch(void* const* d_in, const int* in_sizes, int n_in,
                              void* d_out, int out_size, void* d_ws, size_t ws_size,
                              hipStream_t stream) {
    const float* x        = (const float*)d_in[0];   // (B,256,48,48)
    const float* norm_g   = (const float*)d_in[1];
    const float* norm_b   = (const float*)d_in[2];
    const float* in_proj  = (const float*)d_in[3];   // (512,256)
    const float* convx_w  = (const float*)d_in[4];
    const float* convx_b  = (const float*)d_in[5];
    const float* convz_w  = (const float*)d_in[6];
    const float* convz_b  = (const float*)d_in[7];
    const float* xproj_w  = (const float*)d_in[8];   // (48,256)
    const float* dtproj_w = (const float*)d_in[9];   // (256,16)
    const float* dtproj_b = (const float*)d_in[10];
    const float* imp_w    = (const float*)d_in[11];  // (1,256)
    const float* m_inp_w  = (const float*)d_in[12];  // (1024,256)
    const float* m_conv_w = (const float*)d_in[13];  // (512,1,4)
    const float* m_conv_b = (const float*)d_in[14];
    const float* m_xprj_w = (const float*)d_in[15];  // (48,512)
    const float* m_dtw    = (const float*)d_in[16];  // (512,16)
    const float* m_dtb    = (const float*)d_in[17];
    const float* A_log    = (const float*)d_in[18];  // (512,16)
    const float* A_b_log  = (const float*)d_in[19];
    const float* m_D      = (const float*)d_in[20];
    const float* m_out_w  = (const float*)d_in[21];  // (256,512)
    float* out = (float*)d_out;

    const long N = NTOK;
    float* p = (float*)d_ws;
    float* x_norm   = p; p += (long)BATCH * N * DIMC;        // (B,N,256)
    float* xz       = p; p += (long)BATCH * D_INNER * N;     // (B,512,N)
    float* xc       = p; p += (long)BATCH * HALFC * N;       // (B,256,N)
    float* zc       = p; p += (long)BATCH * HALFC * N;       // (B,256,N)
    float* scores   = p; p += (long)BATCH * N;
    float* sr       = p; p += (long)BATCH * N;
    float* x_sorted = p; p += (long)BATCH * N * DIMC;        // (B,N,256)
    float* xz2      = p; p += (long)BATCH * 2 * D_INNER * N; // (B,1024,N)
    float* xcm      = p; p += (long)BATCH * D_INNER * N;     // (B,512,N)
    float* xdm      = p; p += (long)BATCH * 48 * N;          // (B,48,N)
    float* deltam   = p; p += (long)BATCH * D_INNER * N;     // (B,512,N)
    float* y2       = p; p += (long)BATCH * D_INNER * N;     // (B,512,N)

    const int NT = (int)(N / 16);           // 144 N-tiles
    dim3 gblk(32, 4);
    (void)zc; (void)ws_size; (void)n_in; (void)in_sizes; (void)out_size;

    // 1. layernorm + tokenize
    layernorm_tok<<<BATCH * (int)N, DIMC, 0, stream>>>(x, norm_g, norm_b, x_norm);

    // 2. in_proj: xz[b,512,N] = W[512,256] * x_norm^T   (token-major B, 32-row waves)
    wmma_gemm<true, false, 2><<<dim3(NT / 4, 512 / 32, BATCH), gblk, 0, stream>>>(
        in_proj, x_norm, nullptr, xz, 512, (int)N, DIMC, N * DIMC, 512L * N);

    // 3. dwconv+silu on xp / zp (same-pad left=1)
    {
        int total = BATCH * HALFC * (int)N;
        int blocks = (total + 255) / 256;
        dwconv_silu<<<blocks, 256, 0, stream>>>(xz,            xc, convx_w, convx_b,
                                                HALFC, 512L * N, 256L * N, 1, 0, total);
        dwconv_silu<<<blocks, 256, 0, stream>>>(xz + 256L * N, zc, convz_w, convz_b,
                                                HALFC, 512L * N, 256L * N, 1, 0, total);
    }

    // 4. importance scores
    score_kernel<<<BATCH * (int)N, HALFC, 0, stream>>>(xc, xproj_w, dtproj_w,
                                                       dtproj_b, imp_w, scores);

    // 5. soft rank + fractional scatter reorder
    softrank_kernel<<<BATCH * (int)N, 256, 0, stream>>>(scores, sr);
    {
        int total = BATCH * (int)N * DIMC;
        zero_kernel<<<(total + 255) / 256, 256, 0, stream>>>(x_sorted, total);
    }
    scatter_kernel<<<BATCH * (int)N, DIMC, 0, stream>>>(x_norm, sr, x_sorted);

    // 6. m_in_proj: xz2[b,1024,N] = W2[1024,256] * x_sorted^T
    wmma_gemm<true, false, 2><<<dim3(NT / 4, 1024 / 32, BATCH), gblk, 0, stream>>>(
        m_inp_w, x_sorted, nullptr, xz2, 1024, (int)N, DIMC, N * DIMC, 2L * D_INNER * N);

    // 7. two mamba branches (forward rev=0, backward rev=1), sequential buffer reuse
    for (int rev = 0; rev <= 1; ++rev) {
        int total = BATCH * D_INNER * (int)N;
        int blocks = (total + 255) / 256;
        // causal dwconv (pad 3,0) + silu on (possibly flipped) x half of xz2
        dwconv_silu<<<blocks, 256, 0, stream>>>(xz2, xcm, m_conv_w, m_conv_b,
                                                D_INNER, 2L * D_INNER * N,
                                                (long)D_INNER * N, D_CONV - 1, rev, total);
        // xproj: xd[b,48,N] = Wx[48,512] * xcm  (channel-major B, 48-row waves)
        wmma_gemm<false, false, 3><<<dim3(NT / 4, 1, BATCH), gblk, 0, stream>>>(
            m_xprj_w, xcm, nullptr, xdm, 48, (int)N, D_INNER,
            (long)D_INNER * N, 48L * N);
        // delta = softplus(dtproj(dt) + b)
        delta_kernel<<<blocks, 256, 0, stream>>>(xdm, m_dtw, m_dtb, deltam, total);
        // selective scan + gate; forward writes, backward accumulates (flipped)
        scan_kernel<<<dim3(1, D_INNER / 16, BATCH), 256,
                      2 * 4 * 16 * SCAN_T * sizeof(float), stream>>>(
            deltam, xcm, xdm, xz2, rev ? A_b_log : A_log, m_D, y2, rev);
    }

    // 8. out_proj + residual: out[b,256,N] = Wo[256,512] * y2 + x
    wmma_gemm<false, true, 2><<<dim3(NT / 4, 256 / 32, BATCH), gblk, 0, stream>>>(
        m_out_w, y2, x, out, 256, (int)N, D_INNER,
        (long)D_INNER * N, (long)DIMC * N);
}
